// MultiHeadAttention_60464549593346
// MI455X (gfx1250) — matrix operand order repaired and checked by reading, not compiled, not run on hardware
//
#include <hip/hip_runtime.h>
#include <hip/hip_bf16.h>

typedef __attribute__((ext_vector_type(16))) _Float16 v16h;
typedef __attribute__((ext_vector_type(8)))  _Float16 v8h;
typedef __attribute__((ext_vector_type(8)))  float    v8f;
typedef __attribute__((ext_vector_type(4)))  int      v4i;

#define EMB   1024
#define HEADS 16
#define HDIM  64
#define SEQ   2048
#define BATCH 4
#define MTOT  (BATCH * SEQ)   // 8192 token rows

static __device__ __forceinline__ v16h load16v(const _Float16* p0, const _Float16* p1) {
  union { v16h v; v8h h[2]; } u;
  u.h[0] = *(const v8h*)p0;
  u.h[1] = *(const v8h*)p1;
  return u.v;
}

static __device__ __forceinline__ v8f wmma_f16(v16h a, v16h b, v8f c) {
  // D = A(16x32 f16) * B(32x16 f16) + C(16x16 f32)
  return __builtin_amdgcn_wmma_f32_16x16x32_f16(false, a, false, b, (short)0, c, false, false);
}

// ---- CDNA5 async global->LDS copy (16 bytes per lane, ASYNCcnt-tracked) ----
static __device__ __forceinline__ void async_copy16(const _Float16* g, _Float16* l) {
#if __has_builtin(__builtin_amdgcn_global_load_async_to_lds_b128)
  __builtin_amdgcn_global_load_async_to_lds_b128(
      (__attribute__((address_space(1))) v4i*)g,
      (__attribute__((address_space(3))) v4i*)l, 0, 0);
#else
  asm volatile("global_load_async_to_lds_b128 %0, %1, off"
               :: "v"((unsigned)(size_t)l), "v"(g) : "memory");
#endif
}

static __device__ __forceinline__ void wait_async0() {
#if __has_builtin(__builtin_amdgcn_s_wait_asynccnt)
  __builtin_amdgcn_s_wait_asynccnt(0);
#else
  asm volatile("s_wait_asynccnt 0" ::: "memory");
#endif
}

// ---------------- cast x (fp32 -> f16) ----------------
__global__ void cast_f16_kernel(const float* __restrict__ src, _Float16* __restrict__ dst, int n) {
  int i = blockIdx.x * blockDim.x + threadIdx.x;
  int stride = gridDim.x * blockDim.x;
  for (; i < n; i += stride) dst[i] = (_Float16)src[i];
}

// ------------- transpose+cast weight: Wt[o*E+k] = (f16)W[k*E+o] -------------
__global__ void transpose_cast_kernel(const float* __restrict__ W, _Float16* __restrict__ Wt) {
  __shared__ float tile[32][33];
  int k0 = blockIdx.x * 32;
  int o0 = blockIdx.y * 32;
  int tx = threadIdx.x, ty = threadIdx.y;
  for (int i = 0; i < 32; i += 8)
    tile[ty + i][tx] = W[(size_t)(k0 + ty + i) * EMB + (o0 + tx)];
  __syncthreads();
  for (int i = 0; i < 32; i += 8)
    Wt[(size_t)(o0 + ty + i) * EMB + (k0 + tx)] = (_Float16)tile[tx][ty + i];
}

// ---------------- WMMA GEMM: out = A[M,E] @ Wt^T + bias ----------------
// Block: 256 threads (8 waves).  32-row M tile shared by all waves via async
// LDS staging (double buffered).  Each wave owns a 64-col group; 8 accums.
// mode 0: f16 out [b,h,n,d] (Q,K)   mode 1: f16 out [b,h,d,n] (V^T)
// mode 2: f32 out [m,e] (final projection)
__global__ void gemm_wmma_kernel(const _Float16* __restrict__ A,
                                 const _Float16* __restrict__ Bt,
                                 const float* __restrict__ bias,
                                 void* __restrict__ outp, int mode) {
  __shared__ __align__(16) _Float16 abuf[2][32 * 72];   // 32 rows x 64 k, stride 72
  const int lane = threadIdx.x & 31;
  const int wave = threadIdx.x >> 5;
  const int l16  = lane & 15;
  const int hs   = lane >> 4;
  const int rowBase = blockIdx.x * 32;
  const int colBase = (blockIdx.y * 8 + wave) * 64;

  // stage one 32x64 A chunk: 256 threads x 16B = 4KB
  const int srow = threadIdx.x >> 3;
  const int sseg = threadIdx.x & 7;
  const _Float16* agsrc = A + (size_t)(rowBase + srow) * EMB + sseg * 8;
  const int ldoff = srow * 72 + sseg * 8;

  async_copy16(agsrc, &abuf[0][0] + ldoff);       // chunk 0 -> buf 0

  v8f acc[8];
#pragma unroll
  for (int t = 0; t < 8; ++t) acc[t] = (v8f){0.f,0.f,0.f,0.f,0.f,0.f,0.f,0.f};

  for (int ch = 0; ch < 16; ++ch) {
    wait_async0();
    __syncthreads();                              // buf[ch&1] visible to all waves
    if (ch < 15)
      async_copy16(agsrc + (ch + 1) * 64, &abuf[(ch + 1) & 1][0] + ldoff);

    const _Float16* ab = &abuf[ch & 1][0];
#pragma unroll
    for (int ks = 0; ks < 2; ++ks) {
      const int k0 = ks * 32;
      const _Float16* ap0 = ab + l16 * 72 + k0 + hs * 8;
      v16h af0 = load16v(ap0, ap0 + 16);
      const _Float16* ap1 = ab + (16 + l16) * 72 + k0 + hs * 8;
      v16h af1 = load16v(ap1, ap1 + 16);
#pragma unroll
      for (int t = 0; t < 4; ++t) {
        int c = colBase + t * 16 + l16;
        const _Float16* bp = Bt + (size_t)c * EMB + ch * 64 + k0 + hs * 8;
        v16h bf = load16v(bp, bp + 16);
        acc[t]     = wmma_f16(af0, bf, acc[t]);
        acc[4 + t] = wmma_f16(af1, bf, acc[4 + t]);
      }
    }
    __syncthreads();                              // done reading buf[ch&1]
  }

#pragma unroll
  for (int sub = 0; sub < 2; ++sub) {
    const int rbase = rowBase + sub * 16 + hs * 8;
#pragma unroll
    for (int t = 0; t < 4; ++t) {
      v8f a = acc[sub * 4 + t];
      int c = colBase + t * 16 + l16;
      float bv = bias[c];
      if (mode == 0) {
        _Float16* out = (_Float16*)outp;
        int h = c >> 6, d = c & 63;
#pragma unroll
        for (int j = 0; j < 8; ++j) {
          int m = rbase + j;
          int b_ = m >> 11, n_ = m & 2047;
          out[((((size_t)b_ * HEADS + h) * SEQ) + n_) * HDIM + d] = (_Float16)(a[j] + bv);
        }
      } else if (mode == 1) {
        _Float16* out = (_Float16*)outp;
        int h = c >> 6, d = c & 63;
        int b_ = rbase >> 11, n0 = rbase & 2047;
        union { v8h v; _Float16 e[8]; } pk;
#pragma unroll
        for (int j = 0; j < 8; ++j) pk.e[j] = (_Float16)(a[j] + bv);
        *(v8h*)(out + ((((size_t)b_ * HEADS + h) * HDIM) + d) * SEQ + n0) = pk.v;
      } else {
        float* out = (float*)outp;
#pragma unroll
        for (int j = 0; j < 8; ++j)
          out[(size_t)(rbase + j) * EMB + c] = a[j] + bv;
      }
    }
  }
}

// ---------------- flash attention ----------------
// Block: 128 threads (4 waves) all on one (b,h) head, consecutive 16-query
// tiles.  K/V chunks (32 keys) async-staged into shared LDS double buffers.
// Q,K: [bh,n,d] f16 ; Vt: [bh,d,n] f16 ; Oh: [token,emb] f16
__global__ void attn_kernel(const _Float16* __restrict__ Q,
                            const _Float16* __restrict__ K,
                            const _Float16* __restrict__ Vt,
                            _Float16* __restrict__ Oh) {
  __shared__ __align__(16) _Float16 kbuf[2][32 * 72];   // 32 keys x 64 d (stride 72)
  __shared__ __align__(16) _Float16 vbuf[2][64 * 40];   // 64 d x 32 keys (stride 40)
  __shared__ __align__(16) _Float16 pbuf[4][16 * 40];   // per-wave P transpose

  const int lane = threadIdx.x & 31;
  const int wave = threadIdx.x >> 5;
  const int l16  = lane & 15;
  const int hs   = lane >> 4;

  const int bh = blockIdx.x >> 5;            // 64 head-batches
  const int qt = (blockIdx.x & 31) * 4 + wave;

  const _Float16* Qb = Q  + (size_t)bh * SEQ * HDIM;
  const _Float16* Kb = K  + (size_t)bh * SEQ * HDIM;
  const _Float16* Vb = Vt + (size_t)bh * HDIM * SEQ;

  // ---- async stage of one 32-key K+V chunk (block-wide) ----
  // K: 32 rows x 128B  -> 2 units/thread ;  V: 64 rows x 64B -> 2 units/thread
  const int tid = threadIdx.x;
  auto stage = [&](int kc, int buf) {
    int u0 = tid, u1 = tid + 128;
    async_copy16(Kb + (size_t)(kc + (u0 >> 3)) * HDIM + (u0 & 7) * 8,
                 &kbuf[buf][0] + (u0 >> 3) * 72 + (u0 & 7) * 8);
    async_copy16(Kb + (size_t)(kc + (u1 >> 3)) * HDIM + (u1 & 7) * 8,
                 &kbuf[buf][0] + (u1 >> 3) * 72 + (u1 & 7) * 8);
    async_copy16(Vb + (size_t)(u0 >> 2) * SEQ + kc + (u0 & 3) * 8,
                 &vbuf[buf][0] + (u0 >> 2) * 40 + (u0 & 3) * 8);
    async_copy16(Vb + (size_t)(u1 >> 2) * SEQ + kc + (u1 & 3) * 8,
                 &vbuf[buf][0] + (u1 >> 2) * 40 + (u1 & 3) * 8);
  };

  const int qrow = qt * 16 + l16;
  const _Float16* qp = Qb + (size_t)qrow * HDIM;
  v16h aq0 = load16v(qp + hs * 16,      qp + hs * 16 + 8);       // dims 0..31
  v16h aq1 = load16v(qp + 32 + hs * 16, qp + 32 + hs * 16 + 8);  // dims 32..63

  float mrun[8], lrun[8];
#pragma unroll
  for (int j = 0; j < 8; ++j) { mrun[j] = -1e30f; lrun[j] = 0.f; }
  v8f of[4];
#pragma unroll
  for (int t = 0; t < 4; ++t) of[t] = (v8f){0.f,0.f,0.f,0.f,0.f,0.f,0.f,0.f};

  _Float16* P = &pbuf[wave][0];
  const v8f vz = (v8f){0.f,0.f,0.f,0.f,0.f,0.f,0.f,0.f};
  const float sc = 0.03125f;   // 1/sqrt(EMB)

  stage(0, 0);

  for (int c = 0; c < 64; ++c) {
    wait_async0();
    __syncthreads();                         // K/V chunk visible to all waves
    if (c < 63) stage((c + 1) * 32, (c + 1) & 1);

    const _Float16* kb = &kbuf[c & 1][0];
    const _Float16* vb = &vbuf[c & 1][0];

    // ---- S = Q K^T for 32 keys (two 16-key C fragments), K from LDS ----
    const _Float16* kp0 = kb + l16 * 72 + hs * 16;
    v8f s0 = wmma_f16(aq0, load16v(kp0,      kp0 + 8),      vz);
    s0     = wmma_f16(aq1, load16v(kp0 + 32, kp0 + 32 + 8), s0);
    const _Float16* kp1 = kb + (16 + l16) * 72 + hs * 16;
    v8f s1 = wmma_f16(aq0, load16v(kp1,      kp1 + 8),      vz);
    s1     = wmma_f16(aq1, load16v(kp1 + 32, kp1 + 32 + 8), s1);

    // ---- online softmax ----
    float mn[8], al[8];
#pragma unroll
    for (int j = 0; j < 8; ++j) {
      s0[j] *= sc; s1[j] *= sc;
      float t = fmaxf(s0[j], s1[j]);
      t = fmaxf(t, __shfl_xor(t, 8, 16));
      t = fmaxf(t, __shfl_xor(t, 4, 16));
      t = fmaxf(t, __shfl_xor(t, 2, 16));
      t = fmaxf(t, __shfl_xor(t, 1, 16));
      mn[j] = fmaxf(mrun[j], t);
      al[j] = __expf(mrun[j] - mn[j]);
    }
#pragma unroll
    for (int j = 0; j < 8; ++j) {
      s0[j] = __expf(s0[j] - mn[j]);
      s1[j] = __expf(s1[j] - mn[j]);
      float r = s0[j] + s1[j];
      r += __shfl_xor(r, 8, 16);
      r += __shfl_xor(r, 4, 16);
      r += __shfl_xor(r, 2, 16);
      r += __shfl_xor(r, 1, 16);
      lrun[j] = lrun[j] * al[j] + r;
      mrun[j] = mn[j];
    }
#pragma unroll
    for (int t = 0; t < 4; ++t)
#pragma unroll
      for (int j = 0; j < 8; ++j) of[t][j] *= al[j];

    // ---- transpose P (C-layout -> A-layout) via per-wave LDS ----
#pragma unroll
    for (int j = 0; j < 8; ++j) {
      P[(j + 8 * hs) * 40 + l16]      = (_Float16)s0[j];
      P[(j + 8 * hs) * 40 + 16 + l16] = (_Float16)s1[j];
    }
    asm volatile("s_wait_dscnt 0" ::: "memory");
    const _Float16* pr = P + l16 * 40 + hs * 16;
    v16h pa = load16v(pr, pr + 8);

    // ---- O += P V, V from LDS ----
#pragma unroll
    for (int t = 0; t < 4; ++t) {
      const _Float16* vp = vb + (t * 16 + l16) * 40 + hs * 16;
      of[t] = wmma_f16(pa, load16v(vp, vp + 8), of[t]);
    }
    __syncthreads();                         // done with buf[c&1]
  }

  // ---- normalize and store [token, emb] ----
  float inv[8];
#pragma unroll
  for (int j = 0; j < 8; ++j) inv[j] = 1.0f / lrun[j];
  const int b_ = bh >> 4, h = bh & 15;
#pragma unroll
  for (int t = 0; t < 4; ++t) {
    int col = h * 64 + t * 16 + l16;
#pragma unroll
    for (int j = 0; j < 8; ++j) {
      int token = qt * 16 + j + 8 * hs;
      Oh[((size_t)b_ * SEQ + token) * EMB + col] = (_Float16)(of[t][j] * inv[j]);
    }
  }
}

extern "C" void kernel_launch(void* const* d_in, const int* in_sizes, int n_in,
                              void* d_out, int out_size, void* d_ws, size_t ws_size,
                              hipStream_t stream) {
  const float* x  = (const float*)d_in[0];
  const float* Wq = (const float*)d_in[1];
  const float* bq = (const float*)d_in[2];
  const float* Wk = (const float*)d_in[3];
  const float* bk = (const float*)d_in[4];
  const float* Wv = (const float*)d_in[5];
  const float* bv = (const float*)d_in[6];
  const float* Wo = (const float*)d_in[7];
  const float* bo = (const float*)d_in[8];

  _Float16* ws = (_Float16*)d_ws;
  const size_t NX = (size_t)MTOT * EMB;     // 8388608
  const size_t NW = (size_t)EMB * EMB;      // 1048576
  _Float16* xh  = ws;
  _Float16* wqT = xh  + NX;
  _Float16* wkT = wqT + NW;
  _Float16* wvT = wkT + NW;
  _Float16* woT = wvT + NW;
  _Float16* Qb  = woT + NW;
  _Float16* Kb  = Qb  + NX;
  _Float16* Vtb = Kb  + NX;
  _Float16* Ohb = Vtb + NX;

  // 1) casts
  cast_f16_kernel<<<4096, 256, 0, stream>>>(x, xh, (int)NX);
  {
    dim3 g(EMB / 32, EMB / 32), b(32, 8);
    transpose_cast_kernel<<<g, b, 0, stream>>>(Wq, wqT);
    transpose_cast_kernel<<<g, b, 0, stream>>>(Wk, wkT);
    transpose_cast_kernel<<<g, b, 0, stream>>>(Wv, wvT);
    transpose_cast_kernel<<<g, b, 0, stream>>>(Wo, woT);
  }

  // 2) QKV projections (WMMA + async LDS staging)
  {
    dim3 g(MTOT / 32, 2), b(256);
    gemm_wmma_kernel<<<g, b, 0, stream>>>(xh, wqT, bq, (void*)Qb,  0);
    gemm_wmma_kernel<<<g, b, 0, stream>>>(xh, wkT, bk, (void*)Kb,  0);
    gemm_wmma_kernel<<<g, b, 0, stream>>>(xh, wvT, bv, (void*)Vtb, 1);
  }

  // 3) flash attention (WMMA, online softmax, async K/V staging)
  attn_kernel<<<64 * 32, 128, 0, stream>>>(Qb, Kb, Vtb, Ohb);

  // 4) output projection -> fp32 d_out
  {
    dim3 g(MTOT / 32, 2), b(256);
    gemm_wmma_kernel<<<g, b, 0, stream>>>(Ohb, woT, bo, d_out, 2);
  }
}
